// ContextAwareAttention_884763263516
// MI455X (gfx1250) — matrix operand order repaired and checked by reading, not compiled, not run on hardware
//
#include <hip/hip_runtime.h>
#include <hip/hip_bf16.h>

// ---------------------------------------------------------------------------
// Types for CDNA5 WMMA (wave32): 16x16x32 f16 -> f32
// ---------------------------------------------------------------------------
typedef __attribute__((ext_vector_type(16))) _Float16 v16h;
typedef __attribute__((ext_vector_type(8)))  _Float16 v8h;
typedef __attribute__((ext_vector_type(8)))  float    v8f;
typedef unsigned int v4u __attribute__((ext_vector_type(4)));
typedef int          v4i __attribute__((ext_vector_type(4)));
typedef int          v8i __attribute__((ext_vector_type(8)));

#if defined(__has_builtin)
#if __has_builtin(__builtin_amdgcn_tensor_load_to_lds) && \
    __has_builtin(__builtin_amdgcn_s_wait_tensorcnt)
#define HAVE_TDM 1
#endif
#endif
#ifndef HAVE_TDM
#define HAVE_TDM 0
#endif

__device__ __forceinline__ v8f wmma_f16(v16h a, v16h b, v8f c) {
  // 8 args: (neg_a, A, neg_b, B, c_mod, C, reuse_a, reuse_b)
  return __builtin_amdgcn_wmma_f32_16x16x32_f16(false, a, false, b, (short)0, c,
                                                false, false);
}

__device__ __forceinline__ v8f zero8() {
  v8f z;
#pragma unroll
  for (int i = 0; i < 8; ++i) z[i] = 0.f;
  return z;
}

__device__ __forceinline__ v16h ld_frag(const _Float16* p_lo, const _Float16* p_hi) {
  union { v16h f; struct { v8h lo; v8h hi; } h; } u;
  u.h.lo = *(const v8h*)p_lo;   // ds_load_b128
  u.h.hi = *(const v8h*)p_hi;   // ds_load_b128
  return u.f;
}

// A-matrix 16x32 f16 fragment. Lanes 0-15: K0..7 & K16..23; 16-31: K8..15 & K24..31.
__device__ __forceinline__ v16h ldA(const _Float16* row, int sel) {
  return ld_frag(row + sel * 8, row + 16 + sel * 8);
}
// B-matrix 32x16 f16 fragment (tile stored N-major). Lanes 0-15: K0..7 & K16..23; 16-31: K8..15 & K24..31.
__device__ __forceinline__ v16h ldB(const _Float16* col, int sel) {
  return ld_frag(col + sel * 8, col + 16 + sel * 8);
}

// ---------------------------------------------------------------------------
// CDNA5 async memory -> LDS (ASYNCcnt-tracked, no VGPR round trip).
// Low 32 bits of a generic LDS address are the LDS byte offset (ISA 10.2).
// ---------------------------------------------------------------------------
__device__ __forceinline__ void async_copy_b128(_Float16* lds_dst,
                                                const _Float16* gsrc) {
  uint32_t laddr = (uint32_t)(uintptr_t)lds_dst;
  asm volatile("global_load_async_to_lds_b128 %0, %1, off"
               :: "v"(laddr), "v"(gsrc)
               : "memory");
}
__device__ __forceinline__ void async_wait() {
  asm volatile("s_wait_asynccnt 0" ::: "memory");
}

// ---------------------------------------------------------------------------
// Tensor Data Mover: contiguous 1-D tile (8-byte elements), TENSORcnt-tracked.
// D# group0: count=1 | lds_addr | global_addr[56:0] | type=2.
// D# group1: data_size=3 (8B), tensor_dim0 = tile_dim0 = nbytes/8, no padding,
//            no multicast (workgroup_mask=0), no iteration.
// ---------------------------------------------------------------------------
#if HAVE_TDM
__device__ __forceinline__ void tdm_load_contig(void* lds, const void* g,
                                                uint32_t nbytes) {
  uint64_t ga = (uint64_t)(uintptr_t)g;
  uint32_t elems = nbytes >> 3;                 // 8-byte units
  v4u g0;
  g0.x = 1u;                                    // count=1 (valid user D#)
  g0.y = (uint32_t)(uintptr_t)lds;              // lds_addr (byte offset)
  g0.z = (uint32_t)ga;                          // global_addr[31:0]
  g0.w = (uint32_t)(ga >> 32) | 0x80000000u;    // global_addr[56:32] | type=2
  v8i g1;
  g1[0] = (int)(3u << 16);                      // data_size=3 (8B); mask=0
  g1[1] = (int)((elems & 0xFFFFu) << 16);       // tensor_dim0[15:0]
  g1[2] = (int)(elems >> 16);                   // tensor_dim0[31:16]
  g1[3] = (int)((elems & 0xFFFFu) << 16);       // tile_dim0
  g1[4] = 0;                                    // tile_dim1/2 unused
  g1[5] = (int)elems;                           // tensor_dim0_stride[31:0]
  g1[6] = 0;
  g1[7] = 0;
  v4i z4 = {0, 0, 0, 0};
#if __clang_major__ >= 23
  v8i z8 = {0, 0, 0, 0, 0, 0, 0, 0};
  __builtin_amdgcn_tensor_load_to_lds(g0, g1, z4, z4, z8, 0);
#else
  __builtin_amdgcn_tensor_load_to_lds(g0, g1, z4, z4, 0);
#endif
}
#endif

// Stage a contiguous 8KB (64x64 f16) tile into LDS.
__device__ __forceinline__ void stage_8kb(_Float16* lds, const _Float16* g,
                                          int tid, int wave) {
#if HAVE_TDM
  if (wave == 0) tdm_load_contig(lds, g, 8192);
  (void)tid;
#else
#pragma unroll
  for (int i = 0; i < 4; ++i) {                 // 512 v8h units / 128 threads
    int u = tid + i * 128;
    async_copy_b128(lds + u * 8, g + u * 8);
  }
  (void)wave;
#endif
}
__device__ __forceinline__ void stage_wait() {
#if HAVE_TDM
  __builtin_amdgcn_s_wait_tensorcnt(0);
#else
  async_wait();
#endif
}

// ---------------------------------------------------------------------------
// Tiled GEMM: C[M,NOUT] = A[M,K] @ W[K,NOUT]  (W is f32 row-major)
// 256 threads (8 waves), 128x128 tile, K-step 32, double-buffered LDS.
// Wave w -> rows (w>>1)*32 + {0,16}, cols (w&1)*64 + j*16 : 8 WMMAs/k-step.
// EPI 0: f16 store, head-major [B,H=8,N=256,D=64], val *= scale (Q/K/V)
// EPI 1: f32 store, + bias[col]
// EPI 2: f32 store, + bias[col] + addbuf[row,col]  (residual fuse)
// ---------------------------------------------------------------------------
template <bool AHALF, int EPI>
__global__ void __launch_bounds__(256)
gemm128_kernel(const void* __restrict__ Aptr, const float* __restrict__ W,
               const float* __restrict__ bias, const float* __restrict__ addbuf,
               void* __restrict__ outptr, int M, int Kdim, int NOUT, float scale) {
  __shared__ __align__(16) _Float16 As[2][128 * 32];   // row-major [m][k]
  __shared__ __align__(16) _Float16 Bs[2][128 * 32];   // N-major  [n][k]

  const int tid  = threadIdx.x;
  const int lane = tid & 31;
  const int wave = tid >> 5;
  const int m    = lane & 15;
  const int sel  = lane >> 4;
  const int rgrp = wave >> 1;       // 0..3
  const int cgrp = wave & 1;        // 0..1
  const int rowblk = blockIdx.x * 128;
  const int colblk = blockIdx.y * 128;

  auto stageA = [&](int buf, int k0) {
    if (!AHALF) {
      const float* A = (const float*)Aptr;
#pragma unroll
      for (int i = 0; i < 4; ++i) {
        int u = tid + i * 256;            // 1024 float4 units
        int r = u >> 3, c = (u & 7) * 4;
        float4 f = *(const float4*)(A + (size_t)(rowblk + r) * Kdim + k0 + c);
        _Float16* dst = &As[buf][r * 32 + c];
        dst[0] = (_Float16)f.x; dst[1] = (_Float16)f.y;
        dst[2] = (_Float16)f.z; dst[3] = (_Float16)f.w;
      }
    } else {
      const _Float16* A = (const _Float16*)Aptr;
#pragma unroll
      for (int i = 0; i < 2; ++i) {
        int u = tid + i * 256;            // 512 v8h units
        int r = u >> 2, c = (u & 3) * 8;
        async_copy_b128(&As[buf][r * 32 + c],
                        A + (size_t)(rowblk + r) * Kdim + k0 + c);
      }
    }
  };
  auto stageB = [&](int buf, int k0) {
    int n = tid & 127, kr = tid >> 7;
#pragma unroll
    for (int kk = 0; kk < 16; ++kk) {
      int k = kr + kk * 2;
      Bs[buf][n * 32 + k] = (_Float16)W[(size_t)(k0 + k) * NOUT + colblk + n];
    }
  };

  v8f acc[2][4];
#pragma unroll
  for (int r2 = 0; r2 < 2; ++r2)
#pragma unroll
    for (int j = 0; j < 4; ++j) acc[r2][j] = zero8();

  int cur = 0;
  stageA(0, 0);
  stageB(0, 0);
  for (int k0 = 0; k0 < Kdim; k0 += 32) {
    if (AHALF) async_wait();
    __syncthreads();
    int nxt = cur ^ 1;
    if (k0 + 32 < Kdim) {            // stage next slab while computing this one
      stageA(nxt, k0 + 32);
      stageB(nxt, k0 + 32);
    }
    v16h a0 = ldA(&As[cur][(rgrp * 32 + m) * 32], sel);
    v16h a1 = ldA(&As[cur][(rgrp * 32 + 16 + m) * 32], sel);
#pragma unroll
    for (int j = 0; j < 4; ++j) {
      v16h b = ldB(&Bs[cur][(cgrp * 64 + j * 16 + m) * 32], sel);
      acc[0][j] = wmma_f16(a0, b, acc[0][j]);
      acc[1][j] = wmma_f16(a1, b, acc[1][j]);
    }
    cur = nxt;
  }

  // ---- epilogue: C-layout lane (m,sel): row = base + sel*8 + v, col = m ----
#pragma unroll
  for (int r2 = 0; r2 < 2; ++r2) {
    const int row0 = rowblk + rgrp * 32 + r2 * 16 + sel * 8;
#pragma unroll
    for (int j = 0; j < 4; ++j) {
      int col = colblk + cgrp * 64 + j * 16 + m;
#pragma unroll
      for (int v = 0; v < 8; ++v) {
        int row = row0 + v;
        float val = acc[r2][j][v] * scale;
        if constexpr (EPI == 0) {
          int b = row >> 8, n = row & 255, hh = col >> 6, d = col & 63;
          ((_Float16*)outptr)[(((size_t)b * 8 + hh) * 256 + n) * 64 + d] =
              (_Float16)val;
        } else if constexpr (EPI == 1) {
          ((float*)outptr)[(size_t)row * NOUT + col] = val + bias[col];
        } else {
          ((float*)outptr)[(size_t)row * NOUT + col] =
              val + bias[col] + addbuf[(size_t)row * NOUT + col];
        }
      }
    }
  }
}

// ---------------------------------------------------------------------------
// Gather relative position bias to [H=8, N*N] f32
// ---------------------------------------------------------------------------
__global__ void __launch_bounds__(256)
bias_gather_kernel(const float* __restrict__ table, const int* __restrict__ rel,
                   float* __restrict__ biasf) {
  int idx = blockIdx.x * 256 + threadIdx.x;   // over 256*256
  int r = rel[idx];
#pragma unroll
  for (int h = 0; h < 8; ++h)
    biasf[(size_t)h * 65536 + idx] = table[r * 8 + h];
}

// ---------------------------------------------------------------------------
// Fused LayerNorm + ReLU, f32 in -> f16 out. One block per row (512 cols).
// ---------------------------------------------------------------------------
__global__ void __launch_bounds__(256)
ln_relu_kernel(const float* __restrict__ h1, const float* __restrict__ g,
               const float* __restrict__ be, _Float16* __restrict__ out) {
  const int row = blockIdx.x, tid = threadIdx.x;
  const float* r = h1 + (size_t)row * 512;
  float a = r[tid], b = r[tid + 256];
  float s = a + b, ss = a * a + b * b;
#pragma unroll
  for (int off = 1; off < 32; off <<= 1) {
    s  += __shfl_xor(s,  off, 32);
    ss += __shfl_xor(ss, off, 32);
  }
  __shared__ float sh[16];
  int wave = tid >> 5, lane = tid & 31;
  if (lane == 0) { sh[wave] = s; sh[wave + 8] = ss; }
  __syncthreads();
  if (tid == 0) {
    float ts = 0.f, tss = 0.f;
#pragma unroll
    for (int i = 0; i < 8; ++i) { ts += sh[i]; tss += sh[i + 8]; }
    sh[0] = ts; sh[1] = tss;
  }
  __syncthreads();
  float mu  = sh[0] * (1.0f / 512.0f);
  float var = sh[1] * (1.0f / 512.0f) - mu * mu;
  float inv = rsqrtf(var + 1e-5f);
  float y0 = fmaxf((a - mu) * inv * g[tid]       + be[tid],       0.f);
  float y1 = fmaxf((b - mu) * inv * g[tid + 256] + be[tid + 256], 0.f);
  out[(size_t)row * 512 + tid]       = (_Float16)y0;
  out[(size_t)row * 512 + tid + 256] = (_Float16)y1;
}

// ---------------------------------------------------------------------------
// Flash-style attention: one workgroup per (b, h, 64-row block).
// 128 threads = 4 waves; each wave owns 16 query rows.
// Q/K tiles staged by the Tensor Data Mover (contiguous 8KB); V transposed
// manually into LDS; K/V double-buffered so DMA overlaps the WMMAs.
// Output: f16 [B, 256, INNER=512] (col = h*64 + d) for the output projection.
// ---------------------------------------------------------------------------
__global__ void __launch_bounds__(128)
attn_kernel(const _Float16* __restrict__ qh, const _Float16* __restrict__ kh,
            const _Float16* __restrict__ vh, const float* __restrict__ biasf,
            const int* __restrict__ mask, _Float16* __restrict__ o_out) {
  __shared__ __align__(16) _Float16 Qs[64 * 64];
  __shared__ __align__(16) _Float16 Ks[2][64 * 64];    // [n][d]
  __shared__ __align__(16) _Float16 Vt[2][64 * 64];    // [d][n] (transposed)
  __shared__ __align__(16) _Float16 Ps[4][16 * 64];    // per-wave P tile

  const int tid  = threadIdx.x;
  const int lane = tid & 31;
  const int wave = tid >> 5;
  const int m    = lane & 15;
  const int sel  = lane >> 4;
  const int bh   = blockIdx.x;           // b*8 + h
  const int b    = bh >> 3, h = bh & 7;
  const int rblk = blockIdx.y * 64;

  const _Float16* Qg    = qh + ((size_t)bh * 256 + rblk) * 64;
  const _Float16* Kbase = kh + (size_t)bh * 256 * 64;
  const _Float16* Vbase = vh + (size_t)bh * 256 * 64;

  auto stage_vt = [&](int buf, const _Float16* Vg) {
#pragma unroll
    for (int i = 0; i < 4; ++i) {
      int u = tid + i * 128;
      int r = u >> 3, c = (u & 7) * 8;
      v8h xv = *(const v8h*)(Vg + r * 64 + c);
#pragma unroll
      for (int e = 0; e < 8; ++e) Vt[buf][(c + e) * 64 + r] = xv[e];
    }
  };

  // ---- prologue staging: Q + first K/V block ----
  stage_8kb(Qs, Qg, tid, wave);
  stage_8kb(Ks[0], Kbase, tid, wave);
  stage_vt(0, Vbase);
  stage_wait();
  __syncthreads();

  v16h qa[2];
#pragma unroll
  for (int kk = 0; kk < 2; ++kk)
    qa[kk] = ldA(&Qs[(wave * 16 + m) * 64 + kk * 32], sel);

  v8f oacc[4];
  float mi[8], li[8];
#pragma unroll
  for (int j = 0; j < 4; ++j) oacc[j] = zero8();
#pragma unroll
  for (int v = 0; v < 8; ++v) { mi[v] = -1e30f; li[v] = 0.f; }

  const int row_base = rblk + wave * 16 + sel * 8;

  int cur = 0;
  for (int c0 = 0; c0 < 256; c0 += 64) {
    int nxt = cur ^ 1;
    if (c0 + 64 < 256) {               // stage next block during compute
      stage_8kb(Ks[nxt], Kbase + (size_t)(c0 + 64) * 64, tid, wave);
      stage_vt(nxt, Vbase + (size_t)(c0 + 64) * 64);
    }

    // ---- S = Q K^T (scale folded into Q) ----
    v8f s[4];
#pragma unroll
    for (int j = 0; j < 4; ++j) s[j] = zero8();
#pragma unroll
    for (int kk = 0; kk < 2; ++kk) {
#pragma unroll
      for (int j = 0; j < 4; ++j) {
        v16h kb = ldB(&Ks[cur][(j * 16 + m) * 64 + kk * 32], sel);
        s[j] = wmma_f16(qa[kk], kb, s[j]);
      }
    }

    // ---- bias + mask + online softmax (per C-layout row) ----
#pragma unroll
    for (int v = 0; v < 8; ++v) {
      int row = row_base + v;
      float sv[4];
#pragma unroll
      for (int j = 0; j < 4; ++j) {
        int col = c0 + j * 16 + m;
        float x = s[j][v] + biasf[(size_t)h * 65536 + row * 256 + col];
        if (mask[(size_t)b * 65536 + row * 256 + col] == 0) x = -1e30f;
        sv[j] = x;
      }
      float rm = fmaxf(fmaxf(sv[0], sv[1]), fmaxf(sv[2], sv[3]));
#pragma unroll
      for (int off = 1; off < 16; off <<= 1)
        rm = fmaxf(rm, __shfl_xor(rm, off, 32));
      float newm = fmaxf(mi[v], rm);
      float corr = __expf(mi[v] - newm);
      float psum = 0.f;
#pragma unroll
      for (int j = 0; j < 4; ++j) {
        float p = __expf(sv[j] - newm);
        psum += p;
        Ps[wave][(sel * 8 + v) * 64 + j * 16 + m] = (_Float16)p;
      }
#pragma unroll
      for (int off = 1; off < 16; off <<= 1)
        psum += __shfl_xor(psum, off, 32);
      li[v] = li[v] * corr + psum;
      mi[v] = newm;
#pragma unroll
      for (int j = 0; j < 4; ++j) oacc[j][v] *= corr;
    }
    __syncthreads();   // P visible (C-layout stores -> A-layout loads)

    // ---- O += P @ V ----
#pragma unroll
    for (int kk = 0; kk < 2; ++kk) {
      v16h pa = ldA(&Ps[wave][m * 64 + kk * 32], sel);
#pragma unroll
      for (int j = 0; j < 4; ++j) {
        v16h vb = ldB(&Vt[cur][(j * 16 + m) * 64 + kk * 32], sel);
        oacc[j] = wmma_f16(pa, vb, oacc[j]);
      }
    }

    stage_wait();      // next K (TDM) landed
    __syncthreads();   // and all readers of `cur` are done
    cur = nxt;
  }

  // ---- normalize + store f16 [B, 256, 512] ----
#pragma unroll
  for (int v = 0; v < 8; ++v) {
    float inv = 1.0f / li[v];
    int row = row_base + v;
#pragma unroll
    for (int j = 0; j < 4; ++j) {
      int d = j * 16 + m;
      o_out[((size_t)b * 256 + row) * 512 + h * 64 + d] =
          (_Float16)(oacc[j][v] * inv);
    }
  }
}

// ---------------------------------------------------------------------------
// Launch
// ---------------------------------------------------------------------------
extern "C" void kernel_launch(void* const* d_in, const int* in_sizes, int n_in,
                              void* d_out, int out_size, void* d_ws, size_t ws_size,
                              hipStream_t stream) {
  (void)in_sizes; (void)n_in; (void)out_size; (void)ws_size;

  const float* x          = (const float*)d_in[0];
  const float* context    = (const float*)d_in[1];
  const int*   mask       = (const int*)  d_in[2];
  const float* Wq         = (const float*)d_in[3];
  const float* Wk         = (const float*)d_in[4];
  const float* Wv         = (const float*)d_in[5];
  const float* Wc1        = (const float*)d_in[6];
  const float* bc1        = (const float*)d_in[7];
  const float* ln_g       = (const float*)d_in[8];
  const float* ln_b       = (const float*)d_in[9];
  const float* Wc2        = (const float*)d_in[10];
  const float* bc2        = (const float*)d_in[11];
  const float* Wo         = (const float*)d_in[12];
  const float* bo         = (const float*)d_in[13];
  const float* bias_table = (const float*)d_in[14];
  const int*   rel_index  = (const int*)  d_in[15];
  float* out = (float*)d_out;

  // workspace carve-up
  char* ws = (char*)d_ws;
  size_t off = 0;
  auto take = [&](size_t bytes) {
    char* p = ws + off;
    off += (bytes + 255) & ~(size_t)255;
    return p;
  };
  const size_t ME = (size_t)128 * 256 * 512;           // 16,777,216 elements
  _Float16* qh    = (_Float16*)take(ME * 2);
  _Float16* kh    = (_Float16*)take(ME * 2);
  _Float16* vh    = (_Float16*)take(ME * 2);
  _Float16* of16  = (_Float16*)take(ME * 2);
  _Float16* hf16  = (_Float16*)take(ME * 2);
  float*    h1    = (float*)   take(ME * 4);
  float*    biasf = (float*)   take((size_t)8 * 256 * 256 * 4);

  const int M = 128 * 256, K = 512, NO = 512;
  const dim3 gg(M / 128, NO / 128);
  const float SCALE = 0.125f;   // 64^-0.5

  bias_gather_kernel<<<256, 256, 0, stream>>>(bias_table, rel_index, biasf);

  // Q (scale folded), K, V -> f16 head-major
  gemm128_kernel<false, 0><<<gg, 256, 0, stream>>>(x,       Wq, nullptr, nullptr, qh, M, K, NO, SCALE);
  gemm128_kernel<false, 0><<<gg, 256, 0, stream>>>(context, Wk, nullptr, nullptr, kh, M, K, NO, 1.0f);
  gemm128_kernel<false, 0><<<gg, 256, 0, stream>>>(context, Wv, nullptr, nullptr, vh, M, K, NO, 1.0f);

  // attention
  attn_kernel<<<dim3(1024, 4), 128, 0, stream>>>(qh, kh, vh, biasf, mask, of16);

  // context branch: h1 = context@Wc1 + bc1 -> LN -> ReLU -> f16
  gemm128_kernel<false, 1><<<gg, 256, 0, stream>>>(context, Wc1, bc1, nullptr, h1, M, K, NO, 1.0f);
  ln_relu_kernel<<<M, 256, 0, stream>>>(h1, ln_g, ln_b, hf16);

  // ctx = h@Wc2 + bc2 -> d_out
  gemm128_kernel<true, 1><<<gg, 256, 0, stream>>>(hf16, Wc2, bc2, nullptr, out, M, K, NO, 1.0f);
  // d_out = attn_out@Wo + bo + d_out
  gemm128_kernel<true, 2><<<gg, 256, 0, stream>>>(of16, Wo, bo, out, out, M, K, NO, 1.0f);
}
